// InfiniAttention_40432822125106
// MI455X (gfx1250) — matrix operand order repaired and checked by reading, not compiled, not run on hardware
//
#include <hip/hip_runtime.h>
#include <hip/hip_bf16.h>

#define SEQ_LEN 2048
#define HIDDEN  2048
#define NHEADS  32
#define HDIM    64

typedef __bf16 bf16;
typedef __attribute__((ext_vector_type(16))) __bf16 bf16x16;
typedef __attribute__((ext_vector_type(8)))  __bf16 bf16x8;
typedef __attribute__((ext_vector_type(8)))  float  f32x8;

// Exact pointee type from the builtin's signature: GCC vector of 4 ints.
typedef int v4i_gcc __attribute__((vector_size(16)));
typedef __attribute__((address_space(1))) v4i_gcc as1_v4i;  // global
typedef __attribute__((address_space(3))) v4i_gcc as3_v4i;  // LDS

union BF16Frag {
  bf16x16 v;
  struct { bf16x8 lo, hi; } h;
};

// A-matrix 16x32 bf16 fragment per ISA 7.12.2:
// lanes 0-15: M=lane, elems 0..7 -> K=0..7, elems 8..15 -> K=16..23
// lanes 16-31: M=lane-16, elems 0..7 -> K=8..15, elems 8..15 -> K=24..31
__device__ __forceinline__ bf16x16 load_a_frag(const bf16* base, int lda) {
  const int lane = threadIdx.x & 31;
  const int m    = lane & 15;
  const int kh   = lane >> 4;          // 0 or 1 -> K offset 0 or 8
  const bf16* p = base + m * lda + kh * 8;
  BF16Frag f;
  f.h.lo = *(const bf16x8*)(p);        // K = kh*8 + 0..7
  f.h.hi = *(const bf16x8*)(p + 16);   // K = kh*8 + 16..23
  return f.v;
}

// B-matrix 32x16 bf16 fragment: lane -> N=lane&15, elems 0..7 -> K=8*(lane>>4)+0..7,
// elems 8..15 -> K=16+8*(lane>>4)+0..7. Source layout: bsrc[n*ldb + k].
__device__ __forceinline__ bf16x16 load_b_frag(const bf16* base, int ldb) {
  const int lane = threadIdx.x & 31;
  const int n    = lane & 15;
  const int kb   = (lane >> 4) << 3;
  const bf16* p = base + n * ldb + kb;
  BF16Frag f;
  f.h.lo = *(const bf16x8*)(p);
  f.h.hi = *(const bf16x8*)(p + 16);
  return f.v;
}

__device__ __forceinline__ f32x8 wmma_bf16(bf16x16 a, bf16x16 b, f32x8 c) {
  return __builtin_amdgcn_wmma_f32_16x16x32_bf16(false, a, false, b,
                                                 (short)0, c, false, false);
}

// ---- 16B global -> LDS copy, async (gfx1250 path) when available
__device__ __forceinline__ void copy_b128_to_lds(const bf16* g, bf16* l) {
#if __has_builtin(__builtin_amdgcn_global_load_async_to_lds_b128)
  __builtin_amdgcn_global_load_async_to_lds_b128(
      (as1_v4i*)(g), (as3_v4i*)(l), 0, 0);
#else
  *(bf16x8*)l = *(const bf16x8*)g;   // fallback: through VGPRs (load + ds_store)
#endif
}

__device__ __forceinline__ void wait_lds_copies() {
#if __has_builtin(__builtin_amdgcn_global_load_async_to_lds_b128)
#if __has_builtin(__builtin_amdgcn_s_wait_asynccnt)
  __builtin_amdgcn_s_wait_asynccnt(0);
#else
  asm volatile("s_wait_asynccnt 0" ::: "memory");
#endif
#else
  asm volatile("s_wait_dscnt 0" ::: "memory");
#endif
}

// ---------------------------------------------------------------- casts
__global__ void __launch_bounds__(256) cast_f32_bf16_kernel(
    const float* __restrict__ src, bf16* __restrict__ dst, int n) {
  int i = blockIdx.x * blockDim.x + threadIdx.x;
  int stride = gridDim.x * blockDim.x;
  for (; i < n; i += stride) dst[i] = (bf16)src[i];
}

// ---------------------------------------------------------------- RMSNorm
__global__ void __launch_bounds__(256) rmsnorm_kernel(
    const float* __restrict__ x, const float* __restrict__ w,
    bf16* __restrict__ xn) {
  __shared__ float red[256];
  const int s = blockIdx.x;
  const float* row = x + (size_t)s * HIDDEN;
  float ss = 0.f;
  for (int i = threadIdx.x; i < HIDDEN; i += 256) {
    float v = row[i];
    ss += v * v;
  }
  red[threadIdx.x] = ss;
  __syncthreads();
  for (int o = 128; o > 0; o >>= 1) {
    if (threadIdx.x < o) red[threadIdx.x] += red[threadIdx.x + o];
    __syncthreads();
  }
  float inv = rsqrtf(red[0] * (1.0f / HIDDEN) + 1e-5f);
  for (int i = threadIdx.x; i < HIDDEN; i += 256)
    xn[(size_t)s * HIDDEN + i] = (bf16)(row[i] * inv * w[i]);
}

// ---------------------------------------------------------------- GEMM
// C[M x N] = A[M x K](bf16) * W[N x K]^T(bf16), f32 WMMA accumulate.
// Block = 8 waves, tile 128(M) x 64(N). A/B k-slabs (128x32, 64x32) are
// double-buffered in LDS, filled by async global->LDS copies so the copy of
// slab k+1 overlaps the WMMAs on slab k. B is shared by all 8 waves.
__global__ void __launch_bounds__(256) gemm_bf16_kernel(
    const bf16* __restrict__ A, const bf16* __restrict__ W,
    float* __restrict__ C, const float* __restrict__ residual,
    int M, int N, int K) {
  __shared__ __align__(16) bf16 As[2][128 * 32];
  __shared__ __align__(16) bf16 Bs[2][64 * 32];

  const int tid = threadIdx.x;
  const int wib = tid >> 5;
  const int lane = tid & 31;
  const int tilesN = N >> 6;
  const int m0 = (blockIdx.x / tilesN) * 128;
  const int n0 = (blockIdx.x % tilesN) * 64;

  // A slab: 128 rows x 32 bf16 = 512 x 16B chunks -> 2 per thread
  // B slab:  64 rows x 32 bf16 = 256 x 16B chunks -> 1 per thread
  auto issue_copies = [&](int k, int buf) {
#pragma unroll
    for (int i = 0; i < 2; ++i) {
      int c = tid + i * 256;
      int row = c >> 2, seg = (c & 3) * 8;
      copy_b128_to_lds(A + (m0 + row) * K + k + seg, &As[buf][row * 32 + seg]);
    }
    {
      int row = tid >> 2, seg = (tid & 3) * 8;
      copy_b128_to_lds(W + (n0 + row) * K + k + seg, &Bs[buf][row * 32 + seg]);
    }
  };

  issue_copies(0, 0);
  wait_lds_copies();
  __syncthreads();

  f32x8 acc[4] = {};
  int buf = 0;
  for (int k = 0; k < K; k += 32) {
    if (k + 32 < K) issue_copies(k + 32, buf ^ 1);  // overlap with compute

    bf16x16 a = load_a_frag(&As[buf][wib * 16 * 32], 32);
#pragma unroll
    for (int t = 0; t < 4; ++t) {
      bf16x16 b = load_b_frag(&Bs[buf][t * 16 * 32], 32);
      acc[t] = wmma_bf16(a, b, acc[t]);
    }

    wait_lds_copies();   // my copies into buf^1 have landed
    __syncthreads();     // everyone's copies landed; everyone done reading buf
    buf ^= 1;
  }

  // C/D layout: VGPR r -> row +r+8*(lane>=16), col t*16 + (lane&15)
  const int half = lane >> 4;
  const int nc = lane & 15;
  const int mrow = m0 + wib * 16;
#pragma unroll
  for (int t = 0; t < 4; ++t)
#pragma unroll
    for (int r = 0; r < 8; ++r) {
      int row = mrow + r + half * 8;
      int col = n0 + t * 16 + nc;
      float v = acc[t][r];
      if (residual) v += residual[row * N + col];
      C[row * N + col] = v;
    }
}

// ---------------------------------------------------------------- RoPE + split
__global__ void __launch_bounds__(256) rope_split_kernel(
    const float* __restrict__ qkv, bf16* __restrict__ Q,
    bf16* __restrict__ Kb, bf16* __restrict__ Vt) {
  const int s = blockIdx.x;
  const float* row = qkv + (size_t)s * (3 * HIDDEN);
  for (int i = threadIdx.x; i < HIDDEN; i += 256) {
    int h = i >> 6, d = i & 63;
    int dm = d & 31;
    float invf = __powf(10000.0f, -(float)dm * (1.0f / 32.0f));
    float ang = (float)s * invf;
    float c = __cosf(ang), sn = __sinf(ang);
    int pidx = h * 64 + ((d < 32) ? (d + 32) : (d - 32));
    float qv = row[i];
    float kv = row[HIDDEN + i];
    float vv = row[2 * HIDDEN + i];
    float qp = row[pidx];
    float kp = row[HIDDEN + pidx];
    float qr = (d < 32) ? (qv * c - qp * sn) : (qv * c + qp * sn);
    float kr = (d < 32) ? (kv * c - kp * sn) : (kv * c + kp * sn);
    size_t hb = (size_t)h * (SEQ_LEN * HDIM);
    Q[hb + (size_t)s * HDIM + d]  = (bf16)qr;
    Kb[hb + (size_t)s * HDIM + d] = (bf16)kr;
    Vt[hb + (size_t)d * SEQ_LEN + s] = (bf16)vv;
  }
}

// ---------------------------------------------------------------- mem prep
__global__ void __launch_bounds__(256) prep_mem_kernel(
    const float* __restrict__ mk, const float* __restrict__ mv,
    bf16* __restrict__ MK, bf16* __restrict__ MVt) {
  const int h = blockIdx.x;
  const float* kk = mk + h * 4096;
  const float* vv = mv + h * 4096;
  for (int i = threadIdx.x; i < 4096; i += 256) {
    int m = i >> 6, d = i & 63;
    MK[h * 4096 + i] = (bf16)kk[i];
    MVt[h * 4096 + d * 64 + m] = (bf16)vv[i];  // [m][d] -> [d][m]
  }
}

// ---------------------------------------------------------------- attention
// One wave per (head, 16-query tile). Flash-style online softmax over causal
// keys in 32-wide blocks; fused 64-slot memory attention + sigmoid gate.
__global__ void __launch_bounds__(256) attn_kernel(
    const bf16* __restrict__ Q, const bf16* __restrict__ Kb,
    const bf16* __restrict__ Vt, const bf16* __restrict__ MK,
    const bf16* __restrict__ MVt, const float* __restrict__ gate,
    bf16* __restrict__ Ctx) {
  __shared__ __align__(16) bf16 lds_p[8][16 * 32];  // per-wave P staging
  const int wib = threadIdx.x >> 5;
  const int lane = threadIdx.x & 31;
  const int gwave = blockIdx.x * 8 + wib;
  const int head = gwave >> 7;   // 128 q-tiles per head
  const int q0 = (gwave & 127) * 16;
  bf16* pbuf = lds_p[wib];

  const int half = lane >> 4;
  const int nc = lane & 15;
  const float scale = 0.125f;  // 64^-0.5

  const bf16* qbase = Q + (size_t)head * (SEQ_LEN * HDIM) + q0 * HDIM;
  bf16x16 aq0 = load_a_frag(qbase, HDIM);       // K dims 0..31
  bf16x16 aq1 = load_a_frag(qbase + 32, HDIM);  // K dims 32..63

  float m_i[8], l_i[8];
  f32x8 acc[4] = {};
#pragma unroll
  for (int r = 0; r < 8; ++r) { m_i[r] = -3.0e38f; l_i[r] = 0.f; }

  const bf16* kh = Kb + (size_t)head * (SEQ_LEN * HDIM);
  const bf16* vh = Vt + (size_t)head * (HDIM * SEQ_LEN);

  const int jend = q0 + 16;  // causal: keys < jend, wave-uniform bound
  for (int j0 = 0; j0 < jend; j0 += 32) {
    f32x8 st[2];
#pragma unroll
    for (int t = 0; t < 2; ++t) {
      const bf16* kb = kh + (j0 + t * 16) * HDIM;
      f32x8 z = {};
      z = wmma_bf16(aq0, load_b_frag(kb, HDIM), z);
      z = wmma_bf16(aq1, load_b_frag(kb + 32, HDIM), z);
      st[t] = z;
    }
#pragma unroll
    for (int t = 0; t < 2; ++t) {
      int key = j0 + t * 16 + nc;
#pragma unroll
      for (int r = 0; r < 8; ++r) {
        int qrow = q0 + r + half * 8;
        float v = st[t][r] * scale;
        st[t][r] = (key > qrow) ? -3.0e38f : v;
      }
    }
    float alpha[8];
#pragma unroll
    for (int r = 0; r < 8; ++r) {
      float mx = fmaxf(st[0][r], st[1][r]);
#pragma unroll
      for (int off = 1; off < 16; off <<= 1)
        mx = fmaxf(mx, __shfl_xor(mx, off, 32));
      float mnew = fmaxf(m_i[r], mx);
      float a = __expf(m_i[r] - mnew);
      float p0 = __expf(st[0][r] - mnew);
      float p1 = __expf(st[1][r] - mnew);
      st[0][r] = p0; st[1][r] = p1;
      float rs = p0 + p1;
#pragma unroll
      for (int off = 1; off < 16; off <<= 1)
        rs += __shfl_xor(rs, off, 32);
      l_i[r] = l_i[r] * a + rs;
      m_i[r] = mnew;
      alpha[r] = a;
    }
#pragma unroll
    for (int t = 0; t < 4; ++t)
#pragma unroll
      for (int r = 0; r < 8; ++r) acc[t][r] *= alpha[r];

    // P: C/D layout -> A layout via per-wave LDS tile
#pragma unroll
    for (int t = 0; t < 2; ++t)
#pragma unroll
      for (int r = 0; r < 8; ++r)
        pbuf[(r + half * 8) * 32 + t * 16 + nc] = (bf16)st[t][r];
    asm volatile("s_wait_dscnt 0" ::: "memory");
    bf16x16 ap = load_a_frag(pbuf, 32);

    const bf16* vb = vh + j0;
#pragma unroll
    for (int t = 0; t < 4; ++t) {
      bf16x16 bv = load_b_frag(vb + (t * 16) * SEQ_LEN, SEQ_LEN);
      acc[t] = wmma_bf16(ap, bv, acc[t]);
    }
  }

  // memory attention: 64 slots, plain softmax
  f32x8 sm[4];
  const bf16* mkh = MK + head * 4096;
  const bf16* mvh = MVt + head * 4096;
#pragma unroll
  for (int t = 0; t < 4; ++t) {
    const bf16* kb = mkh + (t * 16) * HDIM;
    f32x8 z = {};
    z = wmma_bf16(aq0, load_b_frag(kb, HDIM), z);
    z = wmma_bf16(aq1, load_b_frag(kb + 32, HDIM), z);
    sm[t] = z;
  }
#pragma unroll
  for (int t = 0; t < 4; ++t)
#pragma unroll
    for (int r = 0; r < 8; ++r) sm[t][r] *= scale;
  float msum[8];
#pragma unroll
  for (int r = 0; r < 8; ++r) {
    float mx = fmaxf(fmaxf(sm[0][r], sm[1][r]), fmaxf(sm[2][r], sm[3][r]));
#pragma unroll
    for (int off = 1; off < 16; off <<= 1)
      mx = fmaxf(mx, __shfl_xor(mx, off, 32));
    float rs = 0.f;
#pragma unroll
    for (int t = 0; t < 4; ++t) {
      float p = __expf(sm[t][r] - mx);
      sm[t][r] = p;
      rs += p;
    }
#pragma unroll
    for (int off = 1; off < 16; off <<= 1)
      rs += __shfl_xor(rs, off, 32);
    msum[r] = rs;
  }
  f32x8 macc[4] = {};
#pragma unroll
  for (int blk = 0; blk < 2; ++blk) {
#pragma unroll
    for (int t = 0; t < 2; ++t)
#pragma unroll
      for (int r = 0; r < 8; ++r)
        pbuf[(r + half * 8) * 32 + t * 16 + nc] = (bf16)sm[blk * 2 + t][r];
    asm volatile("s_wait_dscnt 0" ::: "memory");
    bf16x16 ap = load_a_frag(pbuf, 32);
    const bf16* vb = mvh + blk * 32;
#pragma unroll
    for (int t = 0; t < 4; ++t) {
      bf16x16 bv = load_b_frag(vb + (t * 16) * 64, 64);
      macc[t] = wmma_bf16(ap, bv, macc[t]);
    }
  }

  float g = 1.f / (1.f + __expf(-gate[head]));
#pragma unroll
  for (int t = 0; t < 4; ++t)
#pragma unroll
    for (int r = 0; r < 8; ++r) {
      int srow = q0 + r + half * 8;
      int col = head * HDIM + t * 16 + nc;
      float lv = acc[t][r] / l_i[r];
      float mv = macc[t][r] / msum[r];
      Ctx[(size_t)srow * HIDDEN + col] = (bf16)((1.f - g) * lv + g * mv);
    }
}

// ---------------------------------------------------------------- launch
extern "C" void kernel_launch(void* const* d_in, const int* in_sizes, int n_in,
                              void* d_out, int out_size, void* d_ws,
                              size_t ws_size, hipStream_t stream) {
  const float* x      = (const float*)d_in[0];
  const float* qkv_w  = (const float*)d_in[1];
  const float* o_w    = (const float*)d_in[2];
  const float* norm_w = (const float*)d_in[3];
  const float* gate   = (const float*)d_in[4];
  const float* mem_k  = (const float*)d_in[5];
  const float* mem_v  = (const float*)d_in[6];
  float* out = (float*)d_out;

  char* ws = (char*)d_ws;
  size_t off = 0;
  auto alloc = [&](size_t bytes) -> void* {
    void* p = ws + off;
    off += (bytes + 255) & ~(size_t)255;
    return p;
  };
  bf16*  xn   = (bf16*)alloc((size_t)SEQ_LEN * HIDDEN * 2);
  bf16*  wqkv = (bf16*)alloc((size_t)3 * HIDDEN * HIDDEN * 2);
  bf16*  wo   = (bf16*)alloc((size_t)HIDDEN * HIDDEN * 2);
  float* qkv  = (float*)alloc((size_t)SEQ_LEN * 3 * HIDDEN * 4);
  bf16*  Qb   = (bf16*)alloc((size_t)NHEADS * SEQ_LEN * HDIM * 2);
  bf16*  Kb   = (bf16*)alloc((size_t)NHEADS * SEQ_LEN * HDIM * 2);
  bf16*  Vt   = (bf16*)alloc((size_t)NHEADS * SEQ_LEN * HDIM * 2);
  bf16*  MK   = (bf16*)alloc((size_t)NHEADS * 64 * HDIM * 2);
  bf16*  MVt  = (bf16*)alloc((size_t)NHEADS * 64 * HDIM * 2);
  bf16*  Ctx  = (bf16*)alloc((size_t)SEQ_LEN * HIDDEN * 2);

  cast_f32_bf16_kernel<<<1024, 256, 0, stream>>>(qkv_w, wqkv, 3 * HIDDEN * HIDDEN);
  cast_f32_bf16_kernel<<<512, 256, 0, stream>>>(o_w, wo, HIDDEN * HIDDEN);
  rmsnorm_kernel<<<SEQ_LEN, 256, 0, stream>>>(x, norm_w, xn);
  // QKV GEMM: (2048x2048)x(2048x6144): blocks = (2048/128)*(6144/64) = 1536
  gemm_bf16_kernel<<<1536, 256, 0, stream>>>(xn, wqkv, qkv, nullptr,
                                             SEQ_LEN, 3 * HIDDEN, HIDDEN);
  rope_split_kernel<<<SEQ_LEN, 256, 0, stream>>>(qkv, Qb, Kb, Vt);
  prep_mem_kernel<<<NHEADS, 256, 0, stream>>>(mem_k, mem_v, MK, MVt);
  // 32 heads * 128 q-tiles = 4096 waves / 8 = 512 blocks
  attn_kernel<<<512, 256, 0, stream>>>(Qb, Kb, Vt, MK, MVt, gate, Ctx);
  // O-proj + residual: blocks = (2048/128)*(2048/64) = 512
  gemm_bf16_kernel<<<512, 256, 0, stream>>>(Ctx, wo, out, x,
                                            SEQ_LEN, HIDDEN, HIDDEN);
}
